// CrfRnnLayerSPIO_51582557224986
// MI455X (gfx1250) — matrix operand order repaired and checked by reading, not compiled, not run on hardware
//
#include <hip/hip_runtime.h>
#include <hip/hip_bf16.h>

// ---------------------------------------------------------------------------
// CRF-RNN (CrfRnnLayerSPIO) for MI455X / gfx1250, wave32.
//
// Roofline: ~4 GFLOP total, ~100 MB working set -> L2-resident (192 MB);
// bound by VALU stencil work + instruction issue, not HBM (23.3 TB/s).
//  * fused LDS-tiled pass: softmax + 7x7 spatial + bilateral filters with
//    fused normalization (zero padding => norm = sum of in-bounds weights);
//    49 bilateral weights computed once per pixel, reused over 21 channels
//  * channel mixing (3x 21x21 GEMMs over 262144 pixels) on the matrix pipe:
//    v_wmma_f32_16x16x32_f16, M/K padded 21->32, N = 16-pixel column tiles.
//  * all per-pixel channel vectors stored pixel-major with stride 32 so WMMA
//    B operands / u loads / q stores are aligned 32B vector ops, branch-free.
// ---------------------------------------------------------------------------

typedef __attribute__((ext_vector_type(16))) _Float16 v16h;
typedef __attribute__((ext_vector_type(8)))  float    v8f;

constexpr int   C = 21, CP = 32;          // real / padded channel count
constexpr int   H = 512, W = 512, P = H * W;
constexpr int   R = 3, DIAM = 7, NTAP = 49;
constexpr float TA = 80.0f, TB = 13.0f, TG = 3.0f;
constexpr int   SP_IDX = 37;
constexpr int   TSX = 16, TSY = 16;
constexpr int   EX = TSX + 2 * R;         // 22 (tile + halo)
constexpr int   ESZ = EX * EX;            // 484

// ---------------------------------------------------------------- k_init ---
// unaries (1,H,W,21) -> pixel-major padded u/q (stride 32, pads zeroed);
// zero the pad channels of the f16 filter-output buffers once per launch.
__global__ void k_init(const float* __restrict__ un,
                       float* __restrict__ u, float* __restrict__ q,
                       _Float16* __restrict__ so, _Float16* __restrict__ bo) {
    int p = blockIdx.x * blockDim.x + threadIdx.x;
    if (p >= P) return;
    for (int c = 0; c < C; ++c) {
        float v = un[p * C + c];
        u[p * CP + c] = v;
        q[p * CP + c] = v;
    }
    for (int c = C; c < CP; ++c) {
        u[p * CP + c] = 0.f;
        q[p * CP + c] = 0.f;
        so[p * CP + c] = (_Float16)0.f;
        bo[p * CP + c] = (_Float16)0.f;
    }
}

// ------------------------------------------------------------- k_matprep ---
// A1n = -(compat @ Ws), A2n = -(compat @ Wb)   (21x21 each)
__global__ void k_matprep(const float* __restrict__ Ws, const float* __restrict__ Wb,
                          const float* __restrict__ Wc,
                          float* __restrict__ A1n, float* __restrict__ A2n) {
    int t = blockIdx.x * blockDim.x + threadIdx.x;
    if (t >= C * C) return;
    int c = t / C, k = t % C;
    float s1 = 0.f, s2 = 0.f;
    for (int j = 0; j < C; ++j) {
        float wc = Wc[c * C + j];
        s1 += wc * Ws[j * C + k];
        s2 += wc * Wb[j * C + k];
    }
    A1n[t] = -s1;
    A2n[t] = -s2;
}

// -------------------------------------------------------------- k_filter ---
// Fused: softmax(q) on a 22x22 halo tile in LDS, then 7x7 spatial + bilateral
// stencils with fused normalization. Outputs stored f16, pixel-major stride 32
// (pad channels pre-zeroed by k_init and never touched here).
__global__ __launch_bounds__(256) void k_filter(const float* __restrict__ q,
                                                const float* __restrict__ rgb,
                                                _Float16* __restrict__ sout,
                                                _Float16* __restrict__ bout) {
    __shared__ float simg[3 * ESZ];
    __shared__ float ssm[C * ESZ];

    const int tx = threadIdx.x, ty = threadIdx.y;
    const int tid = ty * TSX + tx;
    const int bx0 = blockIdx.x * TSX, by0 = blockIdx.y * TSY;

    // Cooperative halo load: softmax per halo pixel + rgb (already pixel-major).
    for (int idx = tid; idx < ESZ; idx += TSX * TSY) {
        int ly = idx / EX, lx = idx % EX;
        int gy = by0 - R + ly, gx = bx0 - R + lx;
        bool inb = ((unsigned)gy < (unsigned)H) && ((unsigned)gx < (unsigned)W);
        if (inb) {
            int g = gy * W + gx;
            float e[C];
            float m = -1e30f;
            for (int c = 0; c < C; ++c) { e[c] = q[g * CP + c]; m = fmaxf(m, e[c]); }
            float s = 0.f;
            for (int c = 0; c < C; ++c) { e[c] = __expf(e[c] - m); s += e[c]; }
            float inv = 1.f / s;
            for (int c = 0; c < C; ++c) ssm[c * ESZ + idx] = e[c] * inv;
            for (int c = 0; c < 3; ++c) simg[c * ESZ + idx] = rgb[g * 3 + c];
        } else {
            for (int c = 0; c < C; ++c) ssm[c * ESZ + idx] = 0.f;
            for (int c = 0; c < 3; ++c) simg[c * ESZ + idx] = 0.f;
        }
    }
    __syncthreads();

    const int gx = bx0 + tx, gy = by0 + ty, g = gy * W + gx;
    const int l = (ty + R) * EX + (tx + R);

    // Per-pixel tap weights (reused over all 21 channels) + fused norms.
    float cw[NTAP], swg[NTAP];
    float bn = 0.f, sn = 0.f;
    const float r0 = simg[l], g0 = simg[ESZ + l], b0 = simg[2 * ESZ + l];
#pragma unroll
    for (int o = 0; o < NTAP; ++o) {
        int dy = o / DIAM - R, dx = o % DIAM - R;
        int l2 = l + dy * EX + dx;
        bool inb = ((unsigned)(gy + dy) < (unsigned)H) && ((unsigned)(gx + dx) < (unsigned)W);
        float dr = r0 - simg[l2];
        float dg = g0 - simg[ESZ + l2];
        float db = b0 - simg[2 * ESZ + l2];
        float d2 = (float)(dy * dy + dx * dx);
        float sw  = __expf(d2 * (-1.f / (2.f * TA * TA)));
        float cwo = sw * __expf((dr * dr + dg * dg + db * db) * (-1.f / (2.f * TB * TB)));
        float sg  = __expf(d2 * (-1.f / (2.f * TG * TG)));
        cw[o]  = inb ? cwo : 0.f;     // out-of-bounds taps contribute 0 (zero pad)
        swg[o] = sg;                  // sm LDS is 0 out of bounds -> safe
        bn += inb ? cwo : 0.f;
        sn += inb ? sg : 0.f;
    }
    const float isn = 1.f / sn, ibn = 1.f / bn;

    for (int c = 0; c < C; ++c) {
        float as = 0.f, ab = 0.f;
        const float* sc = &ssm[c * ESZ];
#pragma unroll
        for (int o = 0; o < NTAP; ++o) {
            int dy = o / DIAM - R, dx = o % DIAM - R;
            float v = sc[l + dy * EX + dx];
            as += v * swg[o];
            ab += v * cw[o];
        }
        sout[g * CP + c] = (_Float16)(as * isn);
        bout[g * CP + c] = (_Float16)(ab * ibn);
    }
}

// -------------------------------------------------------------- k_update ---
// q_new = u + A1n@sout + A2n@bout + Wsp@(T*q),  T = (q==max_c q)&(seg==37)
// (A1n/A2n are the negated compat-folded matrices.)
// One wave per 16-pixel column tile; M/K padded 21->32 -> two M-blocks, one
// v_wmma_f32_16x16x32_f16 per (matrix, M-block). A operands live in VGPRs
// across 8 tiles per wave; B operands / u / q are aligned 32B vector ops.
__global__ __launch_bounds__(256) void k_update(const float* q_in, float* q_out,
                                                const float* __restrict__ u,
                                                const _Float16* __restrict__ sout,
                                                const _Float16* __restrict__ bout,
                                                const int* __restrict__ segs,
                                                const float* __restrict__ A1n,
                                                const float* __restrict__ A2n,
                                                const float* __restrict__ Wsp) {
    const int lane = threadIdx.x & 31;
    const int wave = blockIdx.x * (blockDim.x >> 5) + (threadIdx.x >> 5);
    const int half = lane >> 4;           // 0: lanes 0-15, 1: lanes 16-31

    // A operand (16x32 f16), k order chosen here to match B: lanes 0-15 row M=lane, halves
    // 0..15 -> K=0..15; lanes 16-31: K=16..31.
    v16h a1[2], a2[2], asp[2];
    const int mrow  = lane & 15;
    const int kbase = half ? 8 : 0;
    for (int m = 0; m < 2; ++m) {
        int M = m * 16 + mrow;
#pragma unroll
        for (int h = 0; h < 16; ++h) {
            int K = half * 16 + h;
            bool ok = (M < C) && (K < C);
            a1[m][h]  = ok ? (_Float16)A1n[M * C + K] : (_Float16)0.f;
            a2[m][h]  = ok ? (_Float16)A2n[M * C + K] : (_Float16)0.f;
            asp[m][h] = ok ? (_Float16)Wsp[M * C + K] : (_Float16)0.f;
        }
    }

    constexpr int TPW = 8;                // tiles (of 16 pixels) per wave
    const int col_n = lane & 15;
    for (int t = 0; t < TPW; ++t) {
        int p0  = (wave * TPW + t) * 16;
        int col = p0 + col_n;

        if (t + 1 < TPW) {                // prefetch next tile's B operands
            __builtin_prefetch(sout + (col + 16) * CP, 0, 1);
            __builtin_prefetch(bout + (col + 16) * CP, 0, 1);
        }

        // B operand (32x16 f16), k order chosen here to match A: lane -> column N=lane&15, halves h ->
        // K = half*16 + h. Pixel-major stride-32 buffers (pads zero) make this
        // one aligned 32-byte vector load per lane per matrix.
        v16h bs = *(const v16h*)(sout + col * CP + half * 16);
        v16h bb = *(const v16h*)(bout + col * CP + half * 16);

        float qa[C];
        float qmax = -1e30f;
        for (int k = 0; k < C; ++k) { qa[k] = q_in[col * CP + k]; qmax = fmaxf(qmax, qa[k]); }
        bool spc = (segs[col] == SP_IDX);

        v16h bsp;
#pragma unroll
        for (int h = 0; h < 16; ++h) {
            int k = half * 16 + h;        // h compile-time: h<5 always in range
            bool kin = (h < 5) || !half;
            float qv = kin ? qa[(h < 5) ? k : (half ? 5 : k)] : 0.f;  // safe idx
            if (kin) qv = qa[k];
            bsp[h] = (kin && spc && (qv == qmax)) ? (_Float16)qv : (_Float16)0.f;
        }

        v8f acc0 = {}; v8f acc1 = {};
        acc0 = __builtin_amdgcn_wmma_f32_16x16x32_f16(false, asp[0], false, bsp, (short)0, acc0, false, false);
        acc0 = __builtin_amdgcn_wmma_f32_16x16x32_f16(false, a1[0],  false, bs,  (short)0, acc0, false, false);
        acc0 = __builtin_amdgcn_wmma_f32_16x16x32_f16(false, a2[0],  false, bb,  (short)0, acc0, false, false);
        acc1 = __builtin_amdgcn_wmma_f32_16x16x32_f16(false, asp[1], false, bsp, (short)0, acc1, false, false);
        acc1 = __builtin_amdgcn_wmma_f32_16x16x32_f16(false, a1[1],  false, bs,  (short)0, acc1, false, false);
        acc1 = __builtin_amdgcn_wmma_f32_16x16x32_f16(false, a2[1],  false, bb,  (short)0, acc1, false, false);

        // C/D layout (16x16 f32): lane -> column N=lane&15, VGPR v -> row
        // M = v + half*8 (+16 for the second M-block). Rows M>=21 of A are
        // zero => acc==0 there, u pads are 0 => q pads stay 0. Branch-free
        // aligned 32B vector load/store pairs.
        const int rbase = half * 8;
        v8f u0 = *(const v8f*)(u + col * CP + rbase);
        v8f u1 = *(const v8f*)(u + col * CP + 16 + rbase);
        *(v8f*)(q_out + col * CP + rbase)      = u0 + acc0;
        *(v8f*)(q_out + col * CP + 16 + rbase) = u1 + acc1;
    }
}

// ----------------------------------------------------------------- k_out ---
// pixel-major padded q -> (1,H,W,21) output.
__global__ void k_out(const float* __restrict__ q, float* __restrict__ out) {
    int idx = blockIdx.x * blockDim.x + threadIdx.x;
    if (idx >= P * C) return;
    int p = idx / C, c = idx % C;
    out[idx] = q[p * CP + c];
}

// ---------------------------------------------------------------------------
extern "C" void kernel_launch(void* const* d_in, const int* in_sizes, int n_in,
                              void* d_out, int out_size, void* d_ws, size_t ws_size,
                              hipStream_t stream) {
    const float* un   = (const float*)d_in[0];   // unaries  (1,H,W,21)
    const float* rgb  = (const float*)d_in[1];   // rgb      (1,H,W,3)
    const int*   segs = (const int*)  d_in[2];   // segs     (1,H,W,1)
    const float* Ws   = (const float*)d_in[3];   // spatial_ker_weights     21x21
    const float* Wb   = (const float*)d_in[4];   // bilateral_ker_weights   21x21
    const float* Wsp  = (const float*)d_in[5];   // superpixel_ker_weights  21x21
    // d_in[6] containment_ker_weights: unused by the reference forward pass
    const float* Wc   = (const float*)d_in[7];   // compatibility_matrix    21x21
    float* out = (float*)d_out;

    // Workspace carve-up (~100 MB, L2-resident): pixel-major stride-32 buffers.
    char* w = (char*)d_ws;
    float*    u   = (float*)w;      w += sizeof(float)    * (size_t)CP * P;
    float*    q   = (float*)w;      w += sizeof(float)    * (size_t)CP * P;
    _Float16* so  = (_Float16*)w;   w += sizeof(_Float16) * (size_t)CP * P;
    _Float16* bo  = (_Float16*)w;   w += sizeof(_Float16) * (size_t)CP * P;
    float*    A1n = (float*)w;      w += sizeof(float) * C * C;
    float*    A2n = (float*)w;      w += sizeof(float) * C * C;

    k_init<<<(P + 255) / 256, 256, 0, stream>>>(un, u, q, so, bo);
    k_matprep<<<(C * C + 255) / 256, 256, 0, stream>>>(Ws, Wb, Wc, A1n, A2n);

    dim3 fgrid(W / TSX, H / TSY), fblk(TSX, TSY);
    // update: (P/16)=16384 column tiles, 8 waves/block * 8 tiles/wave = 64/block
    const int ublocks = (P / 16) / 64;   // 256
    for (int it = 0; it < 3; ++it) {
        k_filter<<<fgrid, fblk, 0, stream>>>(q, rgb, so, bo);
        k_update<<<ublocks, 256, 0, stream>>>(q, q, u, so, bo, segs, A1n, A2n, Wsp);
    }
    k_out<<<(P * C + 255) / 256, 256, 0, stream>>>(q, out);
}
